// Attention_4612794876668
// MI455X (gfx1250) — matrix operand order repaired and checked by reading, not compiled, not run on hardware
//
#include <hip/hip_runtime.h>
#include <hip/hip_bf16.h>
#include <math.h>

typedef __attribute__((ext_vector_type(16))) _Float16     v16h;
typedef __attribute__((ext_vector_type(8)))  float        v8f;
typedef __attribute__((ext_vector_type(4)))  unsigned int u32x4;
typedef __attribute__((ext_vector_type(8)))  int          i32x8;
typedef __attribute__((ext_vector_type(4)))  int          i32x4;

#define BATCH 8
#define DIMC  192
#define QKV_C 576
#define HEADS 8
#define CH    24
#define IMG_W 128
#define HW    16384   // 128*128

#define KTILE 32
#define NTILE 128

#if defined(__gfx1250__) && __has_builtin(__builtin_amdgcn_tensor_load_to_lds)
#define HAVE_TDM 1
#else
#define HAVE_TDM 0
#endif

__device__ __forceinline__ v8f wmma_f16(v16h a, v16h b, v8f c) {
  // D = A(16x32 f16) * B(32x16 f16) + C(16x16 f32)
  return __builtin_amdgcn_wmma_f32_16x16x32_f16(
      /*neg_a=*/false, a, /*neg_b=*/false, b,
      /*c_mod=*/(short)0, c, /*reuse_a=*/false, /*reuse_b=*/false);
}

#if HAVE_TDM
// Issue one TDM 2D tile load: rows x cols f32 elements, row stride
// `stride_elems`, from `gsrc` into LDS at `lds_dst`. Uniform args; caller
// restricts execution to one wave (EXEC is ignored by tensor ops).
__device__ __forceinline__ void tdm_issue_tile(const float* gsrc, void* lds_dst,
                                               unsigned rows, unsigned cols,
                                               unsigned stride_elems,
                                               unsigned tensor_rows) {
  const unsigned long long ga  = (unsigned long long)gsrc;
  const unsigned           lds = (unsigned)(unsigned long long)lds_dst; // low 32 = LDS offset

  // D# group 0: count=1 | lds_addr | global_addr[56:0] | type=2 ("image")
  u32x4 g0 = {
      1u,
      lds,
      (unsigned)(ga & 0xffffffffu),
      (unsigned)((ga >> 32) & 0x01ffffffu) | (2u << 30)
  };
  // D# group 1 bit-packing (see ISA 8.4):
  //  [17:16]  data_size = 2 (4 bytes)
  //  [79:48]  tensor_dim0        [111:80] tensor_dim1
  //  [127:112] tile_dim0         [143:128] tile_dim1
  //  [207:160] tensor_dim0_stride
  const unsigned td0 = stride_elems;   // logical tensor width
  const unsigned td1 = tensor_rows;    // logical tensor height
  i32x8 g1 = {
      (int)(2u << 16),
      (int)(td0 << 16),
      (int)((td0 >> 16) | (td1 << 16)),
      (int)((td1 >> 16) | (cols << 16)),
      (int)rows,
      (int)stride_elems,
      0,
      0
  };
  i32x4 z4 = {0, 0, 0, 0};
#if defined(__clang_major__) && (__clang_major__ >= 23)
  i32x8 z8 = {0, 0, 0, 0, 0, 0, 0, 0};
  __builtin_amdgcn_tensor_load_to_lds(g0, g1, z4, z4, z8, 0);
#else
  __builtin_amdgcn_tensor_load_to_lds(g0, g1, z4, z4, 0);
#endif
}
#endif

// ---------------------------------------------------------------------------
// 1x1 conv as GEMM:  Y[b,m,p] = sum_k Wm[m,k] * X[b,k,p] + bias[m]
// Block = 256 threads (8 waves), C-tile 64(M) x 128(N).
// X tiles (KTILE x NTILE f32) are staged into LDS by the Tensor Data Mover,
// double-buffered, synchronized with s_wait_tensorcnt + workgroup barrier.
// ---------------------------------------------------------------------------
__global__ void conv1x1_wmma_kernel(const float* __restrict__ X,
                                    const float* __restrict__ Wm,
                                    const float* __restrict__ bias,
                                    float* __restrict__ Y,
                                    int M, int K) {
  __shared__ float tileB[2][KTILE][NTILE];   // 32 KB double buffer

  const int b     = blockIdx.z;
  const int m_blk = blockIdx.y * 64;
  const int n_blk = blockIdx.x * NTILE;
  const int tid   = threadIdx.x;
  const int wave  = tid >> 5;
  const int lane  = tid & 31;
  const int lm    = lane & 15;
  const int lh    = lane >> 4;
  const int wm    = wave & 3;
  const int wn    = wave >> 2;
  const int m0    = m_blk + wm * 16;

  const float* Xb = X + (size_t)b * K * HW;
  float*       Yb = Y + (size_t)b * M * HW;
  const int    nk = K / KTILE;

  v8f acc[4] = {};

  // stage tile kb into LDS buffer buf
#if HAVE_TDM
#define STAGE(kb_, buf_)                                                     \
  do {                                                                       \
    if (wave == 0)                                                           \
      tdm_issue_tile(Xb + (size_t)(kb_) * KTILE * HW + n_blk,                \
                     &tileB[(buf_)][0][0], KTILE, NTILE, HW, (unsigned)K);   \
  } while (0)
#else
#define STAGE(kb_, buf_)                                                     \
  do {                                                                       \
    const float* g = Xb + (size_t)(kb_) * KTILE * HW + n_blk;                \
    for (int idx = tid; idx < KTILE * NTILE; idx += 256) {                   \
      const int kk = idx >> 7, nn = idx & (NTILE - 1);                       \
      tileB[(buf_)][kk][nn] = g[(size_t)kk * HW + nn];                       \
    }                                                                        \
  } while (0)
#endif

  STAGE(0, 0);

  for (int kb = 0; kb < nk; ++kb) {
    const int k0 = kb * KTILE;
    if (kb + 1 < nk) {
      STAGE(kb + 1, (kb + 1) & 1);
#if HAVE_TDM
      if (wave == 0) __builtin_amdgcn_s_wait_tensorcnt(1);  // prev tile done
#endif
    } else {
#if HAVE_TDM
      if (wave == 0) __builtin_amdgcn_s_wait_tensorcnt(0);
#endif
    }
    __syncthreads();

    // A fragment: weight rows m0..m0+15, K slice k0..k0+31 (L2-resident)
    v16h a;
    const float* wrow = Wm + (size_t)(m0 + lm) * K + k0;
#pragma unroll
    for (int i = 0; i < 16; ++i) {
      const int kk = (i < 8) ? (lh * 8 + i) : (16 + lh * 8 + (i - 8));
      a[i] = (_Float16)wrow[kk];
    }

    const float(*tb)[NTILE] = tileB[kb & 1];
#pragma unroll
    for (int j = 0; j < 4; ++j) {
      const int nl = wn * 64 + j * 16 + lm;   // local column in tile
      v16h bf;
#pragma unroll
      for (int i = 0; i < 16; ++i)
        bf[i] = (_Float16)tb[(i < 8) ? (lh * 8 + i) : (16 + lh * 8 + (i - 8))][nl];
      acc[j] = wmma_f16(a, bf, acc[j]);
    }
    __syncthreads();
  }
#undef STAGE

#pragma unroll
  for (int j = 0; j < 4; ++j) {
    const int n0 = n_blk + wn * 64 + j * 16;
#pragma unroll
    for (int r = 0; r < 8; ++r) {
      const int m = m0 + lh * 8 + r;
      Yb[(size_t)m * HW + n0 + lm] = acc[j][r] + bias[m];
    }
  }
}

// ---------------------------------------------------------------------------
// Depthwise 3x3, SAME padding. mode 0: out = conv+bias ; mode 1: out = gelu(conv)
// mode 2: out += conv.  Input channel c lives at X[(b*c_stride + c_off + c)*HW].
// ---------------------------------------------------------------------------
__global__ void dwconv3x3_kernel(const float* __restrict__ X,
                                 const float* __restrict__ Wd,
                                 const float* __restrict__ bias,
                                 float* __restrict__ Y,
                                 int c_stride, int c_off, int mode) {
  const int C  = gridDim.y;
  const int c  = blockIdx.y;
  const int b  = blockIdx.z;
  const int p  = blockIdx.x * blockDim.x + threadIdx.x;
  const int py = p >> 7;
  const int px = p & (IMG_W - 1);

  const float* src = X + ((size_t)b * c_stride + c_off + c) * HW;
  float wv[9];
#pragma unroll
  for (int i = 0; i < 9; ++i) wv[i] = Wd[c * 9 + i];

  float s = 0.f;
#pragma unroll
  for (int ky = -1; ky <= 1; ++ky) {
    const int yy = py + ky;
    if (yy < 0 || yy > IMG_W - 1) continue;
#pragma unroll
    for (int kx = -1; kx <= 1; ++kx) {
      const int xx = px + kx;
      if (xx < 0 || xx > IMG_W - 1) continue;
      s += wv[(ky + 1) * 3 + (kx + 1)] * src[yy * IMG_W + xx];
    }
  }

  const size_t oidx = ((size_t)b * C + c) * HW + p;
  if (mode == 0)      Y[oidx] = s + bias[c];
  else if (mode == 1) Y[oidx] = 0.5f * s * (1.0f + erff(s * 0.70710678118654752f));
  else                Y[oidx] += s;
}

// ---------------------------------------------------------------------------
// Per-row 1/max(||row||2, 1e-12) over HW for q (c_off=0) or k (c_off=DIMC).
// ---------------------------------------------------------------------------
__global__ void rownorm_kernel(const float* __restrict__ qkv, int c_off,
                               float* __restrict__ inv) {
  __shared__ float red[256];
  const int r = blockIdx.x;            // 0 .. BATCH*DIMC-1
  const int b = r / DIMC;
  const int c = r % DIMC;
  const float* row = qkv + ((size_t)b * QKV_C + c_off + c) * HW;

  float s = 0.f;
  for (int p = threadIdx.x; p < HW; p += 256) {
    const float v = row[p];
    s += v * v;
  }
  red[threadIdx.x] = s;
  __syncthreads();
  for (int off = 128; off > 0; off >>= 1) {
    if (threadIdx.x < off) red[threadIdx.x] += red[threadIdx.x + off];
    __syncthreads();
  }
  if (threadIdx.x == 0)
    inv[r] = 1.0f / fmaxf(sqrtf(red[0]), 1e-12f);
}

// ---------------------------------------------------------------------------
// Channel attention for one (b, head): S = q̂ k̂ᵀ (24x24 padded to 32x32, WMMA
// over K=16384), softmax(*temperature), then out = P · v (WMMA, K=32 one-shot,
// A held in LDS as f16, streaming N=16384).  Block = 128 threads (4 waves).
// ---------------------------------------------------------------------------
__global__ void xca_attention_kernel(const float* __restrict__ qkv,
                                     const float* __restrict__ qinv,
                                     const float* __restrict__ kinv,
                                     const float* __restrict__ temperature,
                                     float* __restrict__ attn_out) {
  __shared__ float    S[32][33];
  __shared__ _Float16 P[32][34];

  const int bh   = blockIdx.x;
  const int b    = bh >> 3;
  const int hd   = bh & 7;
  const int tid  = threadIdx.x;
  const int wave = tid >> 5;
  const int lane = tid & 31;
  const int lm   = lane & 15;
  const int lh   = lane >> 4;

  const float* qbase = qkv + ((size_t)b * QKV_C + 0 * DIMC + hd * CH) * HW;
  const float* kbase = qkv + ((size_t)b * QKV_C + 1 * DIMC + hd * CH) * HW;
  const float* vbase = qkv + ((size_t)b * QKV_C + 2 * DIMC + hd * CH) * HW;
  const float* qi = qinv + b * DIMC + hd * CH;
  const float* ki = kinv + b * DIMC + hd * CH;

  // ---- phase 1: S = q̂ · k̂ᵀ, one 16x16 tile per wave (2x2 grid) ----
  const int m0 = (wave >> 1) * 16;
  const int n0 = (wave & 1) * 16;
  const int ac = m0 + lm;   // q channel for this lane's A rows
  const int bc = n0 + lm;   // k channel for this lane's B cols
  const float  qs   = (ac < CH) ? qi[ac] : 0.f;
  const float  ks   = (bc < CH) ? ki[bc] : 0.f;
  const float* qrow = qbase + (size_t)((ac < CH) ? ac : 0) * HW;
  const float* krow = kbase + (size_t)((bc < CH) ? bc : 0) * HW;

  v8f acc = {};
  for (int e0 = 0; e0 < HW; e0 += 32) {
    __builtin_prefetch(qrow + e0 + 512, 0, 0);   // global_prefetch 16 tiles (512 elems) ahead
    __builtin_prefetch(krow + e0 + 512, 0, 0);
    v16h a, bb;
#pragma unroll
    for (int i = 0; i < 16; ++i) {
      const int ka = (i < 8) ? (lh * 8 + i) : (16 + lh * 8 + (i - 8));
      a[i]  = (_Float16)(qrow[e0 + ka] * qs);
      bb[i] = (_Float16)(krow[e0 + ka] * ks);
    }
    acc = wmma_f16(a, bb, acc);
  }
#pragma unroll
  for (int r = 0; r < 8; ++r)
    S[m0 + lh * 8 + r][n0 + lm] = acc[r];
  __syncthreads();

  // ---- phase 2: row softmax with temperature, write f16 P (zero-padded) ----
  if (tid < 32) {
    const int c = tid;
    if (c < CH) {
      const float t = temperature[hd];
      float mx = -3.4e38f;
      for (int d = 0; d < CH; ++d) mx = fmaxf(mx, S[c][d] * t);
      float sum = 0.f;
      for (int d = 0; d < CH; ++d) sum += expf(S[c][d] * t - mx);
      const float rs = 1.0f / sum;
      for (int d = 0; d < 32; ++d)
        P[c][d] = (d < CH) ? (_Float16)(expf(S[c][d] * t - mx) * rs)
                           : (_Float16)0.f;
    } else {
      for (int d = 0; d < 32; ++d) P[c][d] = (_Float16)0.f;
    }
  }
  __syncthreads();

  // ---- phase 3: out = P · v ; A fragment from LDS, stream N ----
  const int m0p = (wave & 1) * 16;
  v16h pa;
#pragma unroll
  for (int i = 0; i < 16; ++i) {
    const int kk = (i < 8) ? (lh * 8 + i) : (16 + lh * 8 + (i - 8));
    pa[i] = P[m0p + lm][kk];
  }
  float* obase = attn_out + ((size_t)b * DIMC + hd * CH) * HW;
  for (int n0o = (wave >> 1) * 16; n0o < HW; n0o += 32) {
    v16h vb;
#pragma unroll
    for (int i = 0; i < 16; ++i) {
      const int d = (i < 8) ? (lh * 8 + i) : (16 + lh * 8 + (i - 8));
      vb[i] = (d < CH) ? (_Float16)vbase[(size_t)d * HW + n0o + lm]
                       : (_Float16)0.f;
    }
    v8f oacc = {};
    oacc = wmma_f16(pa, vb, oacc);
#pragma unroll
    for (int r = 0; r < 8; ++r) {
      const int c = m0p + lh * 8 + r;
      if (c < CH) obase[(size_t)c * HW + n0o + lm] = oacc[r];
    }
  }
}

// ---------------------------------------------------------------------------
extern "C" void kernel_launch(void* const* d_in, const int* in_sizes, int n_in,
                              void* d_out, int out_size, void* d_ws, size_t ws_size,
                              hipStream_t stream) {
  (void)in_sizes; (void)n_in; (void)out_size; (void)ws_size;

  const float* x      = (const float*)d_in[0];
  const float* w_qkv  = (const float*)d_in[1];
  const float* b_qkv  = (const float*)d_in[2];
  const float* w_dw   = (const float*)d_in[3];
  const float* b_dw   = (const float*)d_in[4];
  const float* w_proj = (const float*)d_in[5];
  const float* b_proj = (const float*)d_in[6];
  const float* w_pos1 = (const float*)d_in[7];
  const float* w_pos2 = (const float*)d_in[8];
  const float* temp   = (const float*)d_in[9];
  float* out = (float*)d_out;

  // workspace carve-up (floats)
  const size_t qkv_elems = (size_t)BATCH * QKV_C * HW;   // 75,497,472
  const size_t out_elems = (size_t)BATCH * DIMC * HW;    // 25,165,824
  float* qkv_lin = (float*)d_ws;
  float* qkvb    = qkv_lin + qkv_elems;
  float* qinv    = qkvb + qkv_elems;
  float* kinv    = qinv + (size_t)BATCH * DIMC;
  float* attn_o  = kinv + (size_t)BATCH * DIMC;
  float* posb    = attn_o + out_elems;

  // 1. qkv 1x1 conv (GEMM 576x192 x 192x16384 per batch), TDM-staged B tiles
  conv1x1_wmma_kernel<<<dim3(HW / 128, QKV_C / 64, BATCH), 256, 0, stream>>>(
      x, w_qkv, b_qkv, qkv_lin, QKV_C, DIMC);
  // 2. depthwise 3x3 + bias on qkv
  dwconv3x3_kernel<<<dim3(HW / 256, QKV_C, BATCH), 256, 0, stream>>>(
      qkv_lin, w_dw, b_dw, qkvb, QKV_C, 0, /*mode=*/0);
  // 3. L2 inv-norms of q and k rows
  rownorm_kernel<<<BATCH * DIMC, 256, 0, stream>>>(qkvb, 0, qinv);
  rownorm_kernel<<<BATCH * DIMC, 256, 0, stream>>>(qkvb, DIMC, kinv);
  // 4. channel attention per (b, head)
  xca_attention_kernel<<<BATCH * HEADS, 128, 0, stream>>>(
      qkvb, qinv, kinv, temp, attn_o);
  // 5. proj 1x1 conv -> d_out (full overwrite)
  conv1x1_wmma_kernel<<<dim3(HW / 128, DIMC / 64, BATCH), 256, 0, stream>>>(
      attn_o, w_proj, b_proj, out, DIMC, DIMC);
  // 6. pos path: gelu(dwconv(v_inp, w_pos1)) -> posb
  dwconv3x3_kernel<<<dim3(HW / 256, DIMC, BATCH), 256, 0, stream>>>(
      qkvb, w_pos1, nullptr, posb, QKV_C, 2 * DIMC, /*mode=*/1);
  // 7. d_out += dwconv(posb, w_pos2)
  dwconv3x3_kernel<<<dim3(HW / 256, DIMC, BATCH), 256, 0, stream>>>(
      posb, w_pos2, nullptr, out, DIMC, 0, /*mode=*/2);
}
